// LinearFlexAttention_19756849562253
// MI455X (gfx1250) — matrix operand order repaired and checked by reading, not compiled, not run on hardware
//
#include <hip/hip_runtime.h>

// ---------------------------------------------------------------------------
// LinearFlexAttention for MI455X (gfx1250, wave32, WMMA + TDM)
//   q = x Wq^T + bq ; k = x Wk^T + bk ; v = x Wv^T + bv
//   P = sigmoid(sigmoid(qk^T / 32) * mask) ;  out = P v
// Sigmoid attention has no row normalization -> context accumulates directly
// in WMMA f32 accumulators while streaming 128-key chunks.
// Matmuls: v_wmma_f32_16x16x32_f16.  Q-tile staged to LDS via TDM
// (tensor_load_to_lds + s_wait_tensorcnt).
// ---------------------------------------------------------------------------

typedef _Float16 v16h __attribute__((ext_vector_type(16)));
typedef _Float16 v8h  __attribute__((ext_vector_type(8)));
typedef float    v8f  __attribute__((ext_vector_type(8)));
typedef unsigned int u32x4 __attribute__((ext_vector_type(4)));
typedef unsigned int u32x8 __attribute__((ext_vector_type(8)));

#define DIM  1024
#define SEQ  4096
#define BAT  2

#define SHUF16(lo, hi) __builtin_shufflevector((lo), (hi), 0,1,2,3,4,5,6,7,8,9,10,11,12,13,14,15)

static __device__ __forceinline__ v8f wmma_f16(v16h a, v16h b, v8f c) {
  return __builtin_amdgcn_wmma_f32_16x16x32_f16(false, a, false, b, (short)0, c, false, false);
}

// --------------------------- f32 -> f16 convert ----------------------------
__global__ void cvt_f16_kernel(const float* __restrict__ src,
                               _Float16* __restrict__ dst, int n) {
  int i = blockIdx.x * blockDim.x + threadIdx.x;
  int stride = gridDim.x * blockDim.x;
  for (; i < n; i += stride) dst[i] = (_Float16)src[i];
}

// ------------------------------ fused QKV GEMM -----------------------------
// One wave -> 16x32 tiles of Q, K and V (A-fragment reused 6x per K-step).
// V is written transposed: Vt[b][d][s], so the later P@V matmul reads
// contiguous B-fragments.
__global__ __launch_bounds__(256)
void qkv_wmma_kernel(const _Float16* __restrict__ xh,
                     const _Float16* __restrict__ wq,
                     const _Float16* __restrict__ wk,
                     const _Float16* __restrict__ wv,
                     const float* __restrict__ bq,
                     const float* __restrict__ bk,
                     const float* __restrict__ bv,
                     _Float16* __restrict__ Qh,
                     _Float16* __restrict__ Kh,
                     _Float16* __restrict__ Vt) {
  const int lane = threadIdx.x & 31;
  const int wave = threadIdx.x >> 5;
  const int tile = blockIdx.x * 8 + wave;       // 512 x 32 = 16384 tiles
  const int ct   = tile & 31;                   // 32 col-pair tiles (D/32)
  const int mt   = tile >> 5;                   // 512 row tiles (B*S/16)
  const int row0 = mt * 16;
  const int col0 = ct * 32;
  const int half = lane >> 4;                   // lane group 0/1
  const int l16  = lane & 15;

  // A fragment: row = l16; lanes 0-15 hold K 0-15, lanes 16-31 hold
  // K 16-31 (matches B fragment order).
  const _Float16* xrow = xh + (size_t)(row0 + l16) * DIM;
  const int ak0 = half * 16;
  // B fragment: column = l16 (W row, since B[d][e] = W[e][d]);
  // lanes 0-15 hold K 0-15, lanes 16-31 hold K 16-31.
  const int c0 = col0 + l16;
  const int c1 = col0 + 16 + l16;
  const int bk0 = half * 16;
  const _Float16* wq0 = wq + (size_t)c0 * DIM + bk0;
  const _Float16* wq1 = wq + (size_t)c1 * DIM + bk0;
  const _Float16* wk0 = wk + (size_t)c0 * DIM + bk0;
  const _Float16* wk1 = wk + (size_t)c1 * DIM + bk0;
  const _Float16* wv0 = wv + (size_t)c0 * DIM + bk0;
  const _Float16* wv1 = wv + (size_t)c1 * DIM + bk0;

  v8f aq0 = {}, aq1 = {}, ak_ = {}, ak1 = {}, av0 = {}, av1 = {};

  for (int d = 0; d < DIM; d += 32) {
    v8h alo = *(const v8h*)(xrow + d + ak0);
    v8h ahi = *(const v8h*)(xrow + d + 8 + ak0);
    v16h a  = SHUF16(alo, ahi);
    aq0 = wmma_f16(a, *(const v16h*)(wq0 + d), aq0);
    aq1 = wmma_f16(a, *(const v16h*)(wq1 + d), aq1);
    ak_ = wmma_f16(a, *(const v16h*)(wk0 + d), ak_);
    ak1 = wmma_f16(a, *(const v16h*)(wk1 + d), ak1);
    av0 = wmma_f16(a, *(const v16h*)(wv0 + d), av0);
    av1 = wmma_f16(a, *(const v16h*)(wv1 + d), av1);
  }

  const float bq0 = bq[c0], bq1 = bq[c1];
  const float bk0f = bk[c0], bk1f = bk[c1];
  const float bv0 = bv[c0], bv1 = bv[c1];

  // C/D layout: VGPR r -> M = r + 8*half, N = l16
#pragma unroll
  for (int r = 0; r < 8; ++r) {
    const int grow = row0 + r + half * 8;
    Qh[(size_t)grow * DIM + c0] = (_Float16)(aq0[r] + bq0);
    Qh[(size_t)grow * DIM + c1] = (_Float16)(aq1[r] + bq1);
    Kh[(size_t)grow * DIM + c0] = (_Float16)(ak_[r] + bk0f);
    Kh[(size_t)grow * DIM + c1] = (_Float16)(ak1[r] + bk1f);
    const int b_ = grow >> 12;                  // / SEQ
    const int s_ = grow & (SEQ - 1);
    Vt[((size_t)b_ * DIM + c0) * SEQ + s_] = (_Float16)(av0[r] + bv0);
    Vt[((size_t)b_ * DIM + c1) * SEQ + s_] = (_Float16)(av1[r] + bv1);
  }
}

// --------------------------- fused sigmoid attention -----------------------
// Block = 32 queries (2 q-tiles), 8 waves, 256 threads.
// Q tile (32x1024 f16 = 64KB) staged into LDS via TDM (8KB slice per wave).
// Per 128-key chunk:
//   stage 1: wave w = key-subtile [w*16, w*16+16): ONE K B-fragment per
//            d-step multiplied against TWO Q A-fragments (64 WMMAs/chunk);
//            K is read exactly once per block.
//   stage 2: wave w = D-slice [w*128, w*128+128): ONE Vt B-fragment per
//            n-tile multiplied against TWO P A-fragments (64 WMMAs/chunk);
//            V is read exactly once per block.
__global__ __launch_bounds__(256)
void attn_wmma_kernel(const _Float16* __restrict__ Qh,
                      const _Float16* __restrict__ Kh,
                      const _Float16* __restrict__ Vt,
                      const float* __restrict__ mask,
                      float* __restrict__ out) {
  __shared__ __align__(128) _Float16 sQ[32 * DIM];   // 64 KB: query tile
  __shared__ __align__(32)  _Float16 sP[32 * 128];   //  8 KB: prob tile

  const int tid  = threadIdx.x;
  const int lane = tid & 31;
  const int wave = tid >> 5;
  const int half = lane >> 4;
  const int l16  = lane & 15;
  const int qb   = blockIdx.x;                 // 128 query blocks per batch
  const int b    = blockIdx.y;
  const int q0   = qb * 32;

  // ---- TDM stage of Q tile: each wave DMAs its 8KB slice into LDS ----
  {
    const int w = __builtin_amdgcn_readfirstlane(wave);
    const unsigned lds_addr =
        (unsigned)(uintptr_t)(&sQ[0]) + (unsigned)w * 8192u;
    const unsigned long long ga =
        (unsigned long long)(uintptr_t)(Qh + ((size_t)b * SEQ + q0) * DIM) +
        (unsigned long long)w * 8192ull;
    // D# group 0: count=1, lds_addr, global_addr, type=2 ("image")
    u32x4 g0;
    g0[0] = 1u;
    g0[1] = lds_addr;
    g0[2] = (unsigned)(ga & 0xffffffffu);
    g0[3] = (unsigned)((ga >> 32) & 0x01ffffffu) | (2u << 30);
    // D# group 1: 1-D tile, data_size=3 (8B), tensor_dim0=tile_dim0=1024,
    // tensor_dim0_stride=1024, no multicast / padding / iteration.
    u32x8 g1;
    g1[0] = 3u << 16;            // data_size = 8 bytes
    g1[1] = 1024u << 16;         // tensor_dim0[15:0]
    g1[2] = 1u << 16;            // tensor_dim0[31:16]=0 | tensor_dim1[15:0]=1
    g1[3] = 1024u << 16;         // tensor_dim1[31:16]=0 | tile_dim0=1024
    g1[4] = 0u;                  // tile_dim1=0 (unused), tile_dim2=0
    g1[5] = 1024u;               // tensor_dim0_stride[31:0]
    g1[6] = 0u;
    g1[7] = 0u;
    asm volatile("tensor_load_to_lds %0, %1" :: "s"(g0), "s"(g1) : "memory");
    __builtin_amdgcn_s_wait_tensorcnt(0);
  }
  __syncthreads();

  v8f acc[2][8];                                // [q-tile][n-tile]
#pragma unroll
  for (int qt = 0; qt < 2; ++qt)
#pragma unroll
    for (int i = 0; i < 8; ++i) acc[qt][i] = (v8f){};

  const _Float16* q0row = sQ + (size_t)l16 * DIM;        // qt0: A row = l16
  const _Float16* q1row = sQ + (size_t)(16 + l16) * DIM; // qt1
  const int ak0 = half * 16;

  for (int kc = 0; kc < SEQ; kc += 128) {
    // ---------------- stage 1: scores -> P in LDS ----------------
    const int key = kc + wave * 16 + l16;           // B column = this key
    const _Float16* krow =
        Kh + ((size_t)b * SEQ + key) * DIM + half * 16;
    v8f s0 = {}, s1 = {};
    for (int d = 0; d < DIM; d += 32) {
      v16h bb = *(const v16h*)(krow + d);
      v16h a0 = SHUF16(*(const v8h*)(q0row + d + ak0),
                       *(const v8h*)(q0row + d + 8 + ak0));
      v16h a1 = SHUF16(*(const v8h*)(q1row + d + ak0),
                       *(const v8h*)(q1row + d + 8 + ak0));
      s0 = wmma_f16(a0, bb, s0);
      s1 = wmma_f16(a1, bb, s1);
    }
#pragma unroll
    for (int r = 0; r < 8; ++r) {
      const int m = r + half * 8;                   // query row within tile
      {
        const float s  = s0[r] * (1.0f / 32.0f);    // 1/sqrt(1024)
        const float p1 = 1.0f / (1.0f + __expf(-s));
        const float mv = mask[((size_t)b * SEQ + (q0 + m)) * SEQ + key];
        const float p2 = 1.0f / (1.0f + __expf(-p1 * mv));
        sP[m * 128 + wave * 16 + l16] = (_Float16)p2;
      }
      {
        const float s  = s1[r] * (1.0f / 32.0f);
        const float p1 = 1.0f / (1.0f + __expf(-s));
        const float mv = mask[((size_t)b * SEQ + (q0 + 16 + m)) * SEQ + key];
        const float p2 = 1.0f / (1.0f + __expf(-p1 * mv));
        sP[(16 + m) * 128 + wave * 16 + l16] = (_Float16)p2;
      }
    }
    __syncthreads();

    // ---------------- stage 2: acc += P @ Vt-slice ----------------
#pragma unroll
    for (int ks = 0; ks < 128; ks += 32) {
      const _Float16* prow0 = sP + l16 * 128 + ks;
      const _Float16* prow1 = sP + (16 + l16) * 128 + ks;
      v16h a0 = SHUF16(*(const v8h*)(prow0 + ak0),
                       *(const v8h*)(prow0 + 8 + ak0));
      v16h a1 = SHUF16(*(const v8h*)(prow1 + ak0),
                       *(const v8h*)(prow1 + 8 + ak0));
#pragma unroll
      for (int nt = 0; nt < 8; ++nt) {
        const int n = wave * 128 + nt * 16 + l16;   // B column = out channel
        const _Float16* vrow =
            Vt + ((size_t)b * DIM + n) * SEQ + kc + ks + half * 16;
        v16h bb = *(const v16h*)(vrow);
        acc[0][nt] = wmma_f16(a0, bb, acc[0][nt]);
        acc[1][nt] = wmma_f16(a1, bb, acc[1][nt]);
      }
    }
    __syncthreads();                                // P reused next chunk
  }

  // Store fp32 context.
#pragma unroll
  for (int qt = 0; qt < 2; ++qt)
#pragma unroll
    for (int nt = 0; nt < 8; ++nt) {
      const int n = wave * 128 + nt * 16 + l16;
#pragma unroll
      for (int r = 0; r < 8; ++r) {
        const int m = qt * 16 + r + half * 8;
        out[((size_t)b * SEQ + (q0 + m)) * DIM + n] = acc[qt][nt][r];
      }
    }
}

// ------------------------------- launcher ----------------------------------
extern "C" void kernel_launch(void* const* d_in, const int* in_sizes, int n_in,
                              void* d_out, int out_size, void* d_ws, size_t ws_size,
                              hipStream_t stream) {
  (void)in_sizes; (void)n_in; (void)out_size; (void)ws_size;

  const float* x    = (const float*)d_in[0];
  const float* mask = (const float*)d_in[1];
  const float* Wq   = (const float*)d_in[2];
  const float* bq   = (const float*)d_in[3];
  const float* Wk   = (const float*)d_in[4];
  const float* bk   = (const float*)d_in[5];
  const float* Wv   = (const float*)d_in[6];
  const float* bv   = (const float*)d_in[7];
  float* out = (float*)d_out;

  // Workspace layout (f16), ~70 MB total.
  char* ws = (char*)d_ws;
  const size_t nX = (size_t)BAT * SEQ * DIM;   // 8,388,608
  const size_t nW = (size_t)DIM * DIM;         // 1,048,576
  _Float16* xh  = (_Float16*)ws;                       ws += nX * 2;
  _Float16* wqh = (_Float16*)ws;                       ws += nW * 2;
  _Float16* wkh = (_Float16*)ws;                       ws += nW * 2;
  _Float16* wvh = (_Float16*)ws;                       ws += nW * 2;
  _Float16* Qh  = (_Float16*)ws;                       ws += nX * 2;
  _Float16* Kh  = (_Float16*)ws;                       ws += nX * 2;
  _Float16* Vt  = (_Float16*)ws;                       ws += nX * 2;

  cvt_f16_kernel<<<2048, 256, 0, stream>>>(x,  xh,  (int)nX);
  cvt_f16_kernel<<<512,  256, 0, stream>>>(Wq, wqh, (int)nW);
  cvt_f16_kernel<<<512,  256, 0, stream>>>(Wk, wkh, (int)nW);
  cvt_f16_kernel<<<512,  256, 0, stream>>>(Wv, wvh, (int)nW);

  // 512 row tiles * 32 col-pair tiles = 16384 wave tiles / 8 waves per block
  qkv_wmma_kernel<<<2048, 256, 0, stream>>>(xh, wqh, wkh, wvh,
                                            bq, bk, bv, Qh, Kh, Vt);

  dim3 agrid(SEQ / 32, BAT);   // 128 x 2
  attn_wmma_kernel<<<agrid, 256, 0, stream>>>(Qh, Kh, Vt, mask, out);
}
